// GraphReasoning_14345190768781
// MI455X (gfx1250) — matrix operand order repaired and checked by reading, not compiled, not run on hardware
//
#include <hip/hip_runtime.h>
#include <hip/hip_bf16.h>
#include <math.h>

typedef __attribute__((ext_vector_type(16))) _Float16 v16h;
typedef __attribute__((ext_vector_type(8)))  float    v8f;
typedef __attribute__((ext_vector_type(8)))  _Float16 h8;

#define PIX    2304      // 48*48
#define NCOLS  18432     // 8 * 2304  (b*n pixels per modality)
#define LDA    40        // LDS row stride (halfs) for A tile, padded
#define LDB    40        // LDS row stride (halfs) for Bt tile, padded
#define MODSTR 2359296   // 2*4*128*2304 floats per modality in d_out

__device__ __forceinline__ float sigf(float x) { return 1.0f / (1.0f + __expf(-x)); }

__device__ __forceinline__ v8f wmma16(v16h a, v16h b, v8f c) {
  return __builtin_amdgcn_wmma_f32_16x16x32_f16(false, a, false, b, (short)0, c, false, false);
}

// A fragment: 16x32 f16, per CDNA5 ISA layout. `base` = first row of the 16-row slice.
// lane<16: M=lane,     halves 0-7 -> K 0-7,  halves 8-15 -> K 16-23
// lane>=16: M=lane-16, halves 0-7 -> K 8-15, halves 8-15 -> K 24-31
__device__ __forceinline__ v16h load_afrag(const _Float16* base, int lane) {
  const _Float16* row = base + (lane & 15) * LDA + ((lane >> 4) * 8);
  h8 lo = *(const h8*)(row);
  h8 hi = *(const h8*)(row + 16);
  v16h a;
#pragma unroll
  for (int i = 0; i < 8; ++i) { a[i] = lo[i]; a[8 + i] = hi[i]; }
  return a;
}

// B fragment: 32x16 f16 from LDS tile stored transposed [col][k]. `base` = first col.
// lane<16: N=lane, halves 0-7 -> K 0-7, 8-15 -> K 16-23 ; lane>=16: N=lane-16, halves 0-7 -> K 8-15, 8-15 -> K 24-31
__device__ __forceinline__ v16h load_bfrag(const _Float16* base, int lane) {
  const _Float16* row = base + (lane & 15) * LDB + ((lane >> 4) * 8);
  h8 lo = *(const h8*)(row);
  h8 hi = *(const h8*)(row + 16);
  v16h b;
#pragma unroll
  for (int i = 0; i < 8; ++i) { b[i] = lo[i]; b[8 + i] = hi[i]; }
  return b;
}

// ---------------------------------------------------------------------------
// Conversion kernels (f32 -> f16 staging)
// ---------------------------------------------------------------------------
__global__ void k_cvt_feat(const float* __restrict__ fr, const float* __restrict__ fd,
                           _Float16* __restrict__ X16) {
  int i = blockIdx.x * blockDim.x + threadIdx.x;
  const int half = 2 * 512 * PIX;            // one modality
  if (i >= 2 * half) return;
  float v = (i < half) ? fr[i] : fd[i - half];
  X16[i] = (_Float16)v;
}

// dst layout: [mod][n][oc][t*512+ci] ; src: ppm_w (n, oc, ci, ky, kx)
__global__ void k_cvt_w(const float* __restrict__ wr, const float* __restrict__ wd,
                        _Float16* __restrict__ W16) {
  int i = blockIdx.x * blockDim.x + threadIdx.x;
  if (i >= 2 * 4 * 128 * 9 * 512) return;
  int ci   = i & 511;
  int t    = (i >> 9) % 9;
  int rest = (i >> 9) / 9;          // (mod*4+nd)*128+oc
  int oc   = rest & 127;
  int nd   = (rest >> 7) & 3;
  int mod  = rest >> 9;
  const float* src = mod ? wd : wr;
  W16[i] = (_Float16)src[((size_t)(nd * 128 + oc) * 512 + ci) * 9 + t];
}

// Wg16[mod][0..98303] = gru_w (3,128,256) flattened ; biasb[mod][0..383] = gru_b
__global__ void k_cvt_gru(const float* __restrict__ gr, const float* __restrict__ gd,
                          _Float16* __restrict__ Wg16,
                          const float* __restrict__ br, const float* __restrict__ bd,
                          float* __restrict__ biasb) {
  int i = blockIdx.x * blockDim.x + threadIdx.x;
  if (i < 2 * 98304) {
    const float* s = (i < 98304) ? gr : gd;
    Wg16[i] = (_Float16)s[i % 98304];
  }
  if (i < 768) {
    const float* s = (i < 384) ? br : bd;
    biasb[i] = s[i % 384];
  }
}

// ---------------------------------------------------------------------------
// PPM dilated conv as implicit GEMM with WMMA.
// grid: x = 36 pixel tiles (64 px), y = 16 combos (mod*8 + b*4 + dila_idx)
// block: 256 threads = 8 waves (4M x 2N); block tile = 128(oc) x 64(pix)
// per-wave tile 32x32 -> 4 independent WMMA accumulator chains.
// K loop = 9 taps x 16 ci-chunks of 32, LDS double-buffered (1 barrier/step).
// ---------------------------------------------------------------------------
__global__ void k_conv_ppm(const _Float16* __restrict__ X16,
                           const _Float16* __restrict__ W16,
                           const float* __restrict__ bnsR, const float* __restrict__ bnbR,
                           const float* __restrict__ bnsD, const float* __restrict__ bnbD,
                           float* __restrict__ out) {
  __shared__ __align__(16) _Float16 sA[2][128 * LDA];
  __shared__ __align__(16) _Float16 sB[2][64 * LDB];

  const int tid  = threadIdx.x;
  const int lane = tid & 31;
  const int wave = tid >> 5;
  const int wm   = wave >> 1;              // 0..3  (M slice of 32)
  const int wn   = wave & 1;               // 0..1  (N slice of 32)
  const int tileN = blockIdx.x;            // 0..35
  const int combo = blockIdx.y;            // mod*8 + b*4 + dn
  const int mod = combo >> 3;
  const int bi8 = combo & 7;               // b*4 + dn
  const int dn  = combo & 3;
  const int dila = 1 << dn;

  const _Float16* Xb = X16 + (size_t)(combo >> 2) * 512 * PIX;        // [ci][pix]
  const _Float16* Wb = W16 + (size_t)(mod * 4 + dn) * 128 * 4608;     // [oc][K]

  // per-thread staging roles
  const int a_oc = tid >> 1, a_part = tid & 1;
  const int b_nn = tid & 63, b_k0 = tid >> 6;      // b_k0 in 0..3
  const int pix0 = tileN * 64 + b_nn;
  const int py = pix0 / 48, px = pix0 % 48;

  auto load_tiles = [&](int kk, int buf) {
    const int t = kk >> 4, cc = kk & 15;
    const int kbase = cc * 32;
    // A tile (128 x 32)
    const _Float16* srcA = Wb + (size_t)a_oc * 4608 + t * 512 + kbase + a_part * 16;
    *(h8*)(sA[buf] + a_oc * LDA + a_part * 16)     = *(const h8*)(srcA);
    *(h8*)(sA[buf] + a_oc * LDA + a_part * 16 + 8) = *(const h8*)(srcA + 8);
    __builtin_prefetch(srcA + 32, 0, 3);           // global_prefetch_b8
    // B tile (32k x 64px) gathered with dilation + zero pad, transposed [px][k]
    const int y = py + dila * ((t / 3) - 1);
    const int x = px + dila * ((t % 3) - 1);
    const bool inb = ((unsigned)y < 48u) && ((unsigned)x < 48u);
    const int sp = y * 48 + x;
#pragma unroll
    for (int j = 0; j < 8; ++j) {
      int k = b_k0 + 4 * j;
      _Float16 v = (_Float16)0.0f;
      if (inb) v = Xb[(size_t)(kbase + k) * PIX + sp];
      sB[buf][b_nn * LDB + k] = v;
    }
  };

  v8f acc[2][2] = {};
  load_tiles(0, 0);
  __syncthreads();
  for (int kk = 0; kk < 144; ++kk) {
    const int cur = kk & 1;
    if (kk + 1 < 144) load_tiles(kk + 1, cur ^ 1);
    v16h a0 = load_afrag(sA[cur] + (wm * 32) * LDA, lane);
    v16h a1 = load_afrag(sA[cur] + (wm * 32 + 16) * LDA, lane);
    v16h b0 = load_bfrag(sB[cur] + (wn * 32) * LDB, lane);
    v16h b1 = load_bfrag(sB[cur] + (wn * 32 + 16) * LDB, lane);
    acc[0][0] = wmma16(a0, b0, acc[0][0]);
    acc[0][1] = wmma16(a0, b1, acc[0][1]);
    acc[1][0] = wmma16(a1, b0, acc[1][0]);
    acc[1][1] = wmma16(a1, b1, acc[1][1]);
    __syncthreads();
  }

  // ---- epilogue: BN scale/bias + ReLU, scatter to d_out ----
  const float* bs = mod ? bnsD : bnsR;
  const float* bb = mod ? bnbD : bnbR;
  float* outF = out + (size_t)mod * MODSTR;
  const int rbase = ((lane >> 4) << 3);
  const int cbase = (lane & 15);
#pragma unroll
  for (int sm = 0; sm < 2; ++sm) {
#pragma unroll
    for (int sn = 0; sn < 2; ++sn) {
      v8f a = acc[sm][sn];
      int pix = tileN * 64 + wn * 32 + sn * 16 + cbase;
#pragma unroll
      for (int r = 0; r < 8; ++r) {
        int oc = wm * 32 + sm * 16 + rbase + r;
        float v = a[r] * bs[dn * 128 + oc] + bb[dn * 128 + oc];
        v = fmaxf(v, 0.0f);
        outF[(size_t)(bi8 * 128 + oc) * PIX + pix] = v;
      }
    }
  }
}

// ---------------------------------------------------------------------------
// Dense GEMM (WMMA): C[M][18432] = A16[M][256] * B16[256][18432] + bias
// grid: x = 288 col tiles (64), y = M/128, z = modality
// block tile 128x64, per-wave 32x32, double-buffered LDS.
// ---------------------------------------------------------------------------
__global__ void k_gemm(const _Float16* __restrict__ A, size_t strideA,
                       const _Float16* __restrict__ B, size_t strideB,
                       const float* __restrict__ bias, int biasStride,
                       float* __restrict__ C, size_t strideC) {
  __shared__ __align__(16) _Float16 sA[2][128 * LDA];
  __shared__ __align__(16) _Float16 sB[2][64 * LDB];

  const int tid  = threadIdx.x;
  const int lane = tid & 31;
  const int wave = tid >> 5;
  const int wm   = wave >> 1;
  const int wn   = wave & 1;
  const int tileN = blockIdx.x;
  const int mblk  = blockIdx.y;
  const int mod   = blockIdx.z;

  const _Float16* Am = A + mod * strideA + (size_t)mblk * 128 * 256;
  const _Float16* Bm = B + mod * strideB;
  float*          Cm = C + mod * strideC;
  const float*    bp = bias + mod * biasStride + mblk * 128;

  const int a_oc = tid >> 1, a_part = tid & 1;
  const int b_k = tid >> 3, b_nb = (tid & 7) * 8;

  auto load_tiles = [&](int kk, int buf) {
    const _Float16* srcA = Am + a_oc * 256 + kk * 32 + a_part * 16;
    *(h8*)(sA[buf] + a_oc * LDA + a_part * 16)     = *(const h8*)(srcA);
    *(h8*)(sA[buf] + a_oc * LDA + a_part * 16 + 8) = *(const h8*)(srcA + 8);
    h8 v = *(const h8*)(Bm + (size_t)(kk * 32 + b_k) * NCOLS + tileN * 64 + b_nb);
#pragma unroll
    for (int q = 0; q < 8; ++q) sB[buf][(b_nb + q) * LDB + b_k] = v[q];
  };

  v8f acc[2][2] = {};
  load_tiles(0, 0);
  __syncthreads();
  for (int kk = 0; kk < 8; ++kk) {
    const int cur = kk & 1;
    if (kk + 1 < 8) load_tiles(kk + 1, cur ^ 1);
    v16h a0 = load_afrag(sA[cur] + (wm * 32) * LDA, lane);
    v16h a1 = load_afrag(sA[cur] + (wm * 32 + 16) * LDA, lane);
    v16h b0 = load_bfrag(sB[cur] + (wn * 32) * LDB, lane);
    v16h b1 = load_bfrag(sB[cur] + (wn * 32 + 16) * LDB, lane);
    acc[0][0] = wmma16(a0, b0, acc[0][0]);
    acc[0][1] = wmma16(a0, b1, acc[0][1]);
    acc[1][0] = wmma16(a1, b0, acc[1][0]);
    acc[1][1] = wmma16(a1, b1, acc[1][1]);
    __syncthreads();
  }

  const int rbase = ((lane >> 4) << 3);
  const int cbase = (lane & 15);
#pragma unroll
  for (int sm = 0; sm < 2; ++sm) {
#pragma unroll
    for (int sn = 0; sn < 2; ++sn) {
      v8f a = acc[sm][sn];
      int col = tileN * 64 + wn * 32 + sn * 16 + cbase;
#pragma unroll
      for (int r = 0; r < 8; ++r) {
        int lrow = wm * 32 + sm * 16 + rbase + r;
        Cm[(size_t)(mblk * 128 + lrow) * NCOLS + col] = a[r] + bp[lrow];
      }
    }
  }
}

// ---------------------------------------------------------------------------
// Pairwise attention gating
// ---------------------------------------------------------------------------
__global__ void k_pa_gate(const float* __restrict__ out, const float* __restrict__ paw,
                          float* __restrict__ g0, float* __restrict__ g1) {
  int i = blockIdx.x * blockDim.x + threadIdx.x;
  if (i >= 8 * PIX) return;
  int bi = i / PIX, p = i % PIX;
  const float* R = out;
  const float* D = out + MODSTR;
  float a0 = 0.0f, a1 = 0.0f;
  for (int c = 0; c < 128; ++c) {
    float r = R[(size_t)(bi * 128 + c) * PIX + p];
    float d = D[(size_t)(bi * 128 + c) * PIX + p];
    a0 += (r - d) * paw[c];
    a1 += (d - r) * paw[128 + c];
  }
  g0[i] = sigf(a0);
  g1[i] = sigf(a1);
}

__global__ void k_pa_update(float* __restrict__ out, const float* __restrict__ g0,
                            const float* __restrict__ g1) {
  int i = blockIdx.x * blockDim.x + threadIdx.x;
  if (i >= MODSTR) return;
  int p = i % PIX;
  int bi = i / (PIX * 128);
  float* R = out;
  float* D = out + MODSTR;
  float r = R[i], d = D[i];
  float gg0 = g0[bi * PIX + p], gg1 = g1[bi * PIX + p];
  R[i] = d * gg1 + r;
  D[i] = r * gg0 + d;
}

// ---------------------------------------------------------------------------
// Graph GRU stages
// ---------------------------------------------------------------------------
__global__ void k_graph_s(const float* __restrict__ out, const float* __restrict__ wwr,
                          const float* __restrict__ wwd, float* __restrict__ sbuf) {
  int i = blockIdx.x * blockDim.x + threadIdx.x;
  if (i >= 2 * 8 * 3 * PIX) return;
  int p   = i % PIX;
  int k   = (i / PIX) % 3;
  int bi  = (i / (PIX * 3)) % 8;
  int mod = i / (PIX * 3 * 8);
  const float* F  = out + (size_t)mod * MODSTR;
  const float* ww = mod ? wwd : wwr;
  int ii  = bi & 3;
  int o   = (k < ii) ? k : k + 1;
  int bio = (bi & ~3) | o;
  float acc = 0.0f;
  for (int c = 0; c < 128; ++c) {
    float a = F[(size_t)(bi  * 128 + c) * PIX + p];
    float b = F[(size_t)(bio * 128 + c) * PIX + p];
    acc += (a - b) * ww[k * 128 + c];
  }
  sbuf[i] = sigf(acc);
}

__global__ void k_graph_hc(const float* __restrict__ out, const float* __restrict__ sbuf,
                           _Float16* __restrict__ hc16) {
  int i = blockIdx.x * blockDim.x + threadIdx.x;
  if (i >= 2 * 8 * 128 * PIX) return;
  int p   = i % PIX;
  int c   = (i / PIX) % 128;
  int bi  = (i / (PIX * 128)) % 8;
  int mod = i / (PIX * 128 * 8);
  const float* F = out + (size_t)mod * MODSTR;
  const float* S = sbuf + (size_t)mod * 8 * 3 * PIX;
  int ii = bi & 3, bibase = bi & ~3;
  float m = 0.0f;
#pragma unroll
  for (int k = 0; k < 3; ++k) {
    int o = (k < ii) ? k : k + 1;
    float fv = F[(size_t)((bibase | o) * 128 + c) * PIX + p];
    int g = (2 * k + c) % 3;                       // gmap[k,c] = (k*128+c)%3
    m += fv * S[(size_t)(bi * 3 + g) * PIX + p];
  }
  _Float16* H = hc16 + (size_t)mod * 256 * NCOLS;
  size_t P = (size_t)bi * PIX + p;
  H[(size_t)c * NCOLS + P]         = (_Float16)m;
  H[(size_t)(128 + c) * NCOLS + P] = (_Float16)F[(size_t)(bi * 128 + c) * PIX + p];
}

__global__ void k_graph_hr(const float* __restrict__ out, const float* __restrict__ zr,
                           const _Float16* __restrict__ hc16, _Float16* __restrict__ hr16) {
  int i = blockIdx.x * blockDim.x + threadIdx.x;
  if (i >= 2 * 8 * 128 * PIX) return;
  int p   = i % PIX;
  int c   = (i / PIX) % 128;
  int bi  = (i / (PIX * 128)) % 8;
  int mod = i / (PIX * 128 * 8);
  size_t P = (size_t)bi * PIX + p;
  const _Float16* H = hc16 + (size_t)mod * 256 * NCOLS;
  _Float16*       R = hr16 + (size_t)mod * 256 * NCOLS;
  float rpre = zr[(size_t)mod * 256 * NCOLS + (size_t)(128 + c) * NCOLS + P];
  float rr = sigf(rpre);
  float f = out[(size_t)mod * MODSTR + (size_t)(bi * 128 + c) * PIX + p];
  R[(size_t)c * NCOLS + P]         = H[(size_t)c * NCOLS + P];    // reuse m
  R[(size_t)(128 + c) * NCOLS + P] = (_Float16)(f * rr);
}

__global__ void k_graph_upd(float* __restrict__ out, const float* __restrict__ zr,
                            const float* __restrict__ cp,
                            const float* __restrict__ gammaR, const float* __restrict__ gammaD) {
  int i = blockIdx.x * blockDim.x + threadIdx.x;
  if (i >= 2 * 8 * 128 * PIX) return;
  int p   = i % PIX;
  int c   = (i / PIX) % 128;
  int bi  = (i / (PIX * 128)) % 8;
  int mod = i / (PIX * 128 * 8);
  size_t P = (size_t)bi * PIX + p;
  float z    = sigf(zr[(size_t)mod * 256 * NCOLS + (size_t)c * NCOLS + P]);
  float cand = tanhf(cp[(size_t)mod * 128 * NCOLS + (size_t)c * NCOLS + P]);
  size_t fi = (size_t)mod * MODSTR + (size_t)(bi * 128 + c) * PIX + p;
  float f = out[fi];
  float hn = f * (1.0f - z) + cand * z;
  float g = mod ? gammaD[0] : gammaR[0];
  out[fi] = hn * g + f;
}

// ---------------------------------------------------------------------------
extern "C" void kernel_launch(void* const* d_in, const int* in_sizes, int n_in,
                              void* d_out, int out_size, void* d_ws, size_t ws_size,
                              hipStream_t stream) {
  (void)in_sizes; (void)n_in; (void)out_size; (void)ws_size;

  const float* feat_rgb = (const float*)d_in[0];
  const float* feat_dep = (const float*)d_in[1];
  const float* ppm_w_r  = (const float*)d_in[2];
  const float* bns_r    = (const float*)d_in[3];
  const float* bnb_r    = (const float*)d_in[4];
  const float* ppm_w_d  = (const float*)d_in[5];
  const float* bns_d    = (const float*)d_in[6];
  const float* bnb_d    = (const float*)d_in[7];
  const float* pa_w     = (const float*)d_in[8];
  const float* wgt_w_r  = (const float*)d_in[9];
  const float* gru_w_r  = (const float*)d_in[10];
  const float* gru_b_r  = (const float*)d_in[11];
  const float* gamma_r  = (const float*)d_in[12];
  const float* wgt_w_d  = (const float*)d_in[13];
  const float* gru_w_d  = (const float*)d_in[14];
  const float* gru_b_d  = (const float*)d_in[15];
  const float* gamma_d  = (const float*)d_in[16];

  float* out = (float*)d_out;

  // --- workspace carve-up (256B aligned) ---
  char* w = (char*)d_ws;
  auto carve = [&](size_t bytes) -> void* {
    void* p = (void*)w;
    w += (bytes + 255) & ~(size_t)255;
    return p;
  };
  _Float16* X16   = (_Float16*)carve((size_t)2 * 2 * 512 * PIX * 2);      // 9.44 MB
  _Float16* W16   = (_Float16*)carve((size_t)2 * 4 * 128 * 4608 * 2);     // 9.44 MB
  _Float16* Wg16  = (_Float16*)carve((size_t)2 * 98304 * 2);              // 384 KB
  float*    biasb = (float*)carve((size_t)2 * 384 * 4);
  float*    sbuf  = (float*)carve((size_t)2 * 8 * 3 * PIX * 4);
  float*    g0    = (float*)carve((size_t)8 * PIX * 4);
  float*    g1    = (float*)carve((size_t)8 * PIX * 4);
  _Float16* hc16  = (_Float16*)carve((size_t)2 * 256 * NCOLS * 2);        // 18.9 MB
  _Float16* hr16  = (_Float16*)carve((size_t)2 * 256 * NCOLS * 2);        // 18.9 MB
  float*    zrpre = (float*)carve((size_t)2 * 256 * NCOLS * 4);           // 37.7 MB
  float*    cpre  = (float*)carve((size_t)2 * 128 * NCOLS * 4);           // 18.9 MB

  const int B = 256;

  // --- stage f16 operands ---
  k_cvt_feat<<<dim3((2 * 2 * 512 * PIX + B - 1) / B), B, 0, stream>>>(feat_rgb, feat_dep, X16);
  k_cvt_w<<<dim3((2 * 4 * 128 * 9 * 512 + B - 1) / B), B, 0, stream>>>(ppm_w_r, ppm_w_d, W16);
  k_cvt_gru<<<dim3((2 * 98304 + B - 1) / B), B, 0, stream>>>(gru_w_r, gru_w_d, Wg16,
                                                             gru_b_r, gru_b_d, biasb);

  // --- PPM dilated convs (implicit GEMM, WMMA) -> d_out holds features ---
  k_conv_ppm<<<dim3(36, 16), B, 0, stream>>>(X16, W16, bns_r, bnb_r, bns_d, bnb_d, out);

  // --- pairwise gating, 2 iterations ---
  for (int it = 0; it < 2; ++it) {
    k_pa_gate<<<dim3((8 * PIX + B - 1) / B), B, 0, stream>>>(out, pa_w, g0, g1);
    k_pa_update<<<dim3((MODSTR + B - 1) / B), B, 0, stream>>>(out, g0, g1);
  }

  // --- graph GRU, 2 iterations (both modalities via grid.z) ---
  const dim3 gElem((2 * 8 * 128 * PIX + B - 1) / B);
  for (int it = 0; it < 2; ++it) {
    k_graph_s<<<dim3((2 * 8 * 3 * PIX + B - 1) / B), B, 0, stream>>>(out, wgt_w_r, wgt_w_d, sbuf);
    k_graph_hc<<<gElem, B, 0, stream>>>(out, sbuf, hc16);
    // z,r GEMM: A = [Wz;Wr] 256x256, M=256
    k_gemm<<<dim3(288, 2, 2), B, 0, stream>>>(Wg16, (size_t)98304, hc16, (size_t)256 * NCOLS,
                                              biasb, 384, zrpre, (size_t)256 * NCOLS);
    k_graph_hr<<<gElem, B, 0, stream>>>(out, zrpre, hc16, hr16);
    // candidate GEMM: A = Wc 128x256, bias = gru_b[2]
    k_gemm<<<dim3(288, 1, 2), B, 0, stream>>>(Wg16 + 65536, (size_t)98304, hr16,
                                              (size_t)256 * NCOLS, biasb + 256, 384,
                                              cpre, (size_t)128 * NCOLS);
    k_graph_upd<<<gElem, B, 0, stream>>>(out, zrpre, cpre, gamma_r, gamma_d);
  }
}
